// Attention_28140625723664
// MI455X (gfx1250) — matrix operand order repaired and checked by reading, not compiled, not run on hardware
//
#include <hip/hip_runtime.h>
#include <hip/hip_bf16.h>

typedef __bf16 bf16;
typedef __attribute__((ext_vector_type(16))) __bf16 v16bf;
typedef __attribute__((ext_vector_type(8)))  __bf16 v8bf;
typedef __attribute__((ext_vector_type(8)))  float   v8f;

#if __has_builtin(__builtin_amdgcn_global_load_async_to_lds_b128)
#define USE_ASYNC_LDS 1
// exact parameter type per hipcc diagnostic: int __vector(4) *
typedef int v4i __attribute__((vector_size(16)));
typedef __attribute__((address_space(1))) v4i* gv4i_ptr;
typedef __attribute__((address_space(3))) v4i* lv4i_ptr;
#endif

// ---------------------------------------------------------------------------
// helpers
// ---------------------------------------------------------------------------
__device__ inline v8f zero8() {
    v8f z = {0.f, 0.f, 0.f, 0.f, 0.f, 0.f, 0.f, 0.f};
    return z;
}

// Build a 16-element bf16 A/B fragment from two 16-byte chunks.
__device__ inline v16bf frag16(const bf16* p0, const bf16* p1) {
    v8bf a = *(const v8bf*)p0;
    v8bf b = *(const v8bf*)p1;
    v16bf r;
#pragma unroll
    for (int i = 0; i < 8; ++i) { r[i] = a[i]; r[i + 8] = b[i]; }
    return r;
}

__device__ inline v8f wmma_bf16(v16bf a, v16bf b, v8f c) {
    // 8 args: (neg_a, A, neg_b, B, c_mod, C, reuse_a, reuse_b)
    return __builtin_amdgcn_wmma_f32_16x16x32_bf16(false, a, false, b,
                                                   (short)0, c, false, false);
}

// 16B global -> LDS copy: async (no VGPR return, ASYNCcnt) when available.
__device__ inline void copy16_g2l(const bf16* g, bf16* l) {
#ifdef USE_ASYNC_LDS
    __builtin_amdgcn_global_load_async_to_lds_b128(
        (gv4i_ptr)(unsigned long long)g,
        (lv4i_ptr)(unsigned int)(unsigned long long)l,
        0, 0);
#else
    *(v8bf*)l = *(const v8bf*)g;
#endif
}

__device__ inline void wait_async_copies() {
#ifdef USE_ASYNC_LDS
#if __has_builtin(__builtin_amdgcn_s_wait_asynccnt)
    __builtin_amdgcn_s_wait_asynccnt(0);
#else
    asm volatile("s_wait_asynccnt 0" ::: "memory");
#endif
#endif
}

// ---------------------------------------------------------------------------
// f32 -> bf16 conversion (8 elements / thread, 32B load / 16B store)
// ---------------------------------------------------------------------------
__global__ void cvt_f32_bf16(const float* __restrict__ in,
                             bf16* __restrict__ out, int n) {
    int i = (blockIdx.x * blockDim.x + threadIdx.x) * 8;
    if (i >= n) return;
    v8f f = *(const v8f*)(in + i);
    v8bf o;
#pragma unroll
    for (int j = 0; j < 8; ++j) o[j] = (bf16)f[j];
    *(v8bf*)(out + i) = o;
}

// ---------------------------------------------------------------------------
// Shared GEMM core: C[m,n] = sum_k A[m,k] * W[n,k]  (both bf16 row-major, K
// contiguous).  WG = 256 threads = 8 waves arranged 4(M) x 2(N).
// WG tile 128x128, wave tile 32x64 -> acc[2][4] of v8f.  K step = 32.
// Tile staging uses GLOBAL_LOAD_ASYNC_TO_LDS_B128 when available.
// ---------------------------------------------------------------------------
#define GEMM_LDT 40                      // padded LDS k-stride (bf16 elems)
#define GEMM_SMEM_BYTES (2 * 128 * GEMM_LDT * 2)

__device__ inline void gemm_tile_core(const bf16* __restrict__ A,
                                      const bf16* __restrict__ W,
                                      int K, int m0, int n0,
                                      char* smem, v8f acc[2][4]) {
    bf16* sA = (bf16*)smem;                           // [128][GEMM_LDT]
    bf16* sB = (bf16*)(smem + 128 * GEMM_LDT * 2);    // [128][GEMM_LDT]

    const int tid  = threadIdx.x;
    const int lane = tid & 31;
    const int wave = tid >> 5;
    const int wm   = wave >> 1;      // 0..3  (M)
    const int wn   = wave & 1;       // 0..1  (N)
    const int lrow = lane & 15;
    const int lhi  = lane >> 4;

#pragma unroll
    for (int a = 0; a < 2; ++a)
#pragma unroll
        for (int b = 0; b < 4; ++b) acc[a][b] = zero8();

    for (int ks = 0; ks < K; ks += 32) {
        // cooperative tile staging: 512 16B-chunks per operand (2 per thread)
#pragma unroll
        for (int it = 0; it < 2; ++it) {
            int c   = tid + it * 256;
            int row = c >> 2;
            int ko  = (c & 3) * 8;
            copy16_g2l(A + (size_t)(m0 + row) * K + ks + ko,
                       sA + row * GEMM_LDT + ko);
            copy16_g2l(W + (size_t)(n0 + row) * K + ks + ko,
                       sB + row * GEMM_LDT + ko);
        }
        wait_async_copies();
        __syncthreads();

        v16bf af[2], bf_[4];
#pragma unroll
        for (int tm = 0; tm < 2; ++tm) {
            const bf16* p = sA + (wm * 32 + tm * 16 + lrow) * GEMM_LDT;
            af[tm] = frag16(p + lhi * 8, p + 16 + lhi * 8);
        }
#pragma unroll
        for (int tn = 0; tn < 4; ++tn) {
            const bf16* p = sB + (wn * 64 + tn * 16 + lrow) * GEMM_LDT;
            bf_[tn] = frag16(p + lhi * 8, p + 16 + lhi * 8);
        }
#pragma unroll
        for (int tm = 0; tm < 2; ++tm)
#pragma unroll
            for (int tn = 0; tn < 4; ++tn)
                acc[tm][tn] = wmma_bf16(af[tm], bf_[tn], acc[tm][tn]);
        __syncthreads();
    }
}

// ---------------------------------------------------------------------------
// QKV projection: [32768,512] x [1536,512]^T, scatter to q/k/v [64][4096][64]
// ---------------------------------------------------------------------------
__global__ void gemm_qkv(const bf16* __restrict__ x,
                         const bf16* __restrict__ wqkv,
                         bf16* __restrict__ qb, bf16* __restrict__ kb,
                         bf16* __restrict__ vb) {
    extern __shared__ char smem[];
    const int K = 512;
    const int m0 = blockIdx.x * 128;
    const int n0 = blockIdx.y * 128;
    v8f acc[2][4];
    gemm_tile_core(x, wqkv, K, m0, n0, smem, acc);

    const int lane = threadIdx.x & 31, wave = threadIdx.x >> 5;
    const int wm = wave >> 1, wn = wave & 1;
    const int lrow = lane & 15, lhi = lane >> 4;
#pragma unroll
    for (int tm = 0; tm < 2; ++tm)
#pragma unroll
        for (int tn = 0; tn < 4; ++tn) {
            int n = n0 + wn * 64 + tn * 16 + lrow;
            int which = n >> 9;          // 0:q 1:k 2:v
            int nn = n & 511;
            int head = nn >> 6, d = nn & 63;
            bf16* dst = (which == 0) ? qb : (which == 1) ? kb : vb;
#pragma unroll
            for (int e = 0; e < 8; ++e) {
                int m = m0 + wm * 32 + tm * 16 + lhi * 8 + e;
                int b = m >> 12, pos = m & 4095;
                dst[((size_t)(b * 8 + head) * 4096 + pos) * 64 + d] =
                    (bf16)acc[tm][tn][e];
            }
        }
}

// ---------------------------------------------------------------------------
// PSP pooling: [64][4096][64] -> [64][288][64] (rows 277..287 zero padded)
// pool sizes 1,2,4,16, exact block means, side = 64
// ---------------------------------------------------------------------------
__global__ void psp_pool(const bf16* __restrict__ src, bf16* __restrict__ dst) {
    const int m = blockIdx.x;
    const bf16* s = src + (size_t)m * 4096 * 64;
    bf16* d = dst + (size_t)m * 288 * 64;
    for (int idx = threadIdx.x; idx < 288 * 64; idx += blockDim.x) {
        int row = idx >> 6;
        int c = idx & 63;
        float val = 0.f;
        if (row < 277) {
            int ps, base;
            if (row == 0)      { ps = 1;  base = 0;  }
            else if (row < 5)  { ps = 2;  base = 1;  }
            else if (row < 21) { ps = 4;  base = 5;  }
            else               { ps = 16; base = 21; }
            int cell = row - base;
            int i = cell / ps, j = cell % ps;
            int rb = 64 / ps;
            float sum = 0.f;
            for (int y = i * rb; y < (i + 1) * rb; ++y)
                for (int xx = j * rb; xx < (j + 1) * rb; ++xx)
                    sum += (float)s[(y * 64 + xx) * 64 + c];
            val = sum / (float)(rb * rb);
        }
        d[(size_t)row * 64 + c] = (bf16)val;
    }
}

// ---------------------------------------------------------------------------
// Fused attention: per (bh, 64-query tile).  block = 128 threads (4 waves,
// 16 query rows each).  LDS: Kp[288][64], VpT[64][288], S f32[64][288],
// P bf16[64][288]  -> 184320 bytes (within 320KB/WG).
// ---------------------------------------------------------------------------
#define ATTN_SMEM_BYTES (288 * 64 * 2 + 64 * 288 * 2 + 64 * 288 * 4 + 64 * 288 * 2)

__global__ void attn_fused(const bf16* __restrict__ q,
                           const bf16* __restrict__ kp,
                           const bf16* __restrict__ vp,
                           bf16* __restrict__ out) {
    extern __shared__ char smem[];
    bf16* sK  = (bf16*)smem;                        // [288][64]
    bf16* sVT = sK + 288 * 64;                      // [64][288] (d-major)
    float* sS = (float*)(sVT + 64 * 288);           // [64][288]
    bf16* sP  = (bf16*)(sS + 64 * 288);             // [64][288]

    const int bh = blockIdx.x;
    const int q0 = blockIdx.y * 64;
    const int tid = threadIdx.x, lane = tid & 31, wave = tid >> 5;
    const int lrow = lane & 15, lhi = lane >> 4;

    const bf16* kpb = kp + (size_t)bh * 288 * 64;
    const bf16* vpb = vp + (size_t)bh * 288 * 64;

    // stage K tiles (contiguous, async to LDS) and V^T (transposed) into LDS
    for (int c = tid; c < 288 * 64 / 8; c += 128) {
        int off = c * 8;
        copy16_g2l(kpb + off, sK + off);
    }
    for (int c = tid; c < 288 * 64; c += 128) {
        int row = c >> 6, dd = c & 63;
        sVT[dd * 288 + row] = vpb[c];
    }
    wait_async_copies();
    __syncthreads();

    // Q fragments for this wave's 16 rows (K = 64 -> two 32-steps)
    const bf16* qrow =
        q + (((size_t)bh * 4096) + q0 + wave * 16 + lrow) * 64;
    v16bf aq0 = frag16(qrow + lhi * 8,      qrow + 16 + lhi * 8);
    v16bf aq1 = frag16(qrow + 32 + lhi * 8, qrow + 48 + lhi * 8);

    const float scale = 0.125f;   // (C/h)^-0.5 = 64^-0.5

    // S = scale * q @ kp^T  -> LDS (18 key tiles of 16)
#pragma unroll
    for (int t = 0; t < 18; ++t) {
        const bf16* kb0 = sK + (t * 16 + lrow) * 64;
        v16bf b0 = frag16(kb0 + lhi * 8,      kb0 + 16 + lhi * 8);
        v16bf b1 = frag16(kb0 + 32 + lhi * 8, kb0 + 48 + lhi * 8);
        v8f acc = zero8();
        acc = wmma_bf16(aq0, b0, acc);
        acc = wmma_bf16(aq1, b1, acc);
#pragma unroll
        for (int e = 0; e < 8; ++e)
            sS[(wave * 16 + lhi * 8 + e) * 288 + t * 16 + lrow] =
                acc[e] * scale;
    }
    __syncthreads();

    // masked softmax over the 277 valid columns, P -> bf16
    if (tid < 64) {
        float* srow = sS + tid * 288;
        bf16* prow = sP + tid * 288;
        float mx = -1e30f;
        for (int j = 0; j < 277; ++j) mx = fmaxf(mx, srow[j]);
        float sum = 0.f;
        for (int j = 0; j < 277; ++j) {
            float e = __expf(srow[j] - mx);
            srow[j] = e;
            sum += e;
        }
        float inv = 1.f / sum;
        for (int j = 0; j < 277; ++j) prow[j] = (bf16)(srow[j] * inv);
        for (int j = 277; j < 288; ++j) prow[j] = (bf16)0.f;
    }
    __syncthreads();

    // O = P @ Vp   (K = 288, 9 steps; 4 output tiles of 16 along d)
    v8f oacc[4];
#pragma unroll
    for (int tn = 0; tn < 4; ++tn) oacc[tn] = zero8();
    for (int ks = 0; ks < 288; ks += 32) {
        const bf16* pr = sP + (wave * 16 + lrow) * 288 + ks;
        v16bf ap = frag16(pr + lhi * 8, pr + 16 + lhi * 8);
#pragma unroll
        for (int tn = 0; tn < 4; ++tn) {
            const bf16* vb = sVT + (tn * 16 + lrow) * 288 + ks;
            v16bf bv = frag16(vb + lhi * 8, vb + 16 + lhi * 8);
            oacc[tn] = wmma_bf16(ap, bv, oacc[tn]);
        }
    }

    // scatter to [B, N, h*hd] bf16
    const int b = bh >> 3, h = bh & 7;
#pragma unroll
    for (int tn = 0; tn < 4; ++tn) {
        int d = tn * 16 + lrow;
#pragma unroll
        for (int e = 0; e < 8; ++e) {
            int pos = q0 + wave * 16 + lhi * 8 + e;
            out[((size_t)b * 4096 + pos) * 512 + h * 64 + d] =
                (bf16)oacc[tn][e];
        }
    }
}

// ---------------------------------------------------------------------------
// Output projection: [32768,512] x [512,512]^T + bias -> f32 d_out
// ---------------------------------------------------------------------------
__global__ void gemm_proj(const bf16* __restrict__ A,
                          const bf16* __restrict__ W,
                          const float* __restrict__ bias,
                          float* __restrict__ out) {
    extern __shared__ char smem[];
    const int K = 512, N = 512;
    const int m0 = blockIdx.x * 128;
    const int n0 = blockIdx.y * 128;
    v8f acc[2][4];
    gemm_tile_core(A, W, K, m0, n0, smem, acc);

    const int lane = threadIdx.x & 31, wave = threadIdx.x >> 5;
    const int wm = wave >> 1, wn = wave & 1;
    const int lrow = lane & 15, lhi = lane >> 4;
#pragma unroll
    for (int tm = 0; tm < 2; ++tm)
#pragma unroll
        for (int tn = 0; tn < 4; ++tn) {
            int n = n0 + wn * 64 + tn * 16 + lrow;
            float bs = bias[n];
#pragma unroll
            for (int e = 0; e < 8; ++e) {
                int m = m0 + wm * 32 + tm * 16 + lhi * 8 + e;
                out[(size_t)m * N + n] = acc[tm][tn][e] + bs;
            }
        }
}

// ---------------------------------------------------------------------------
// launch
// ---------------------------------------------------------------------------
extern "C" void kernel_launch(void* const* d_in, const int* in_sizes, int n_in,
                              void* d_out, int out_size, void* d_ws,
                              size_t ws_size, hipStream_t stream) {
    const float* x      = (const float*)d_in[0];   // [8,4096,512]
    const float* w_qkv  = (const float*)d_in[1];   // [1536,512]
    const float* w_proj = (const float*)d_in[2];   // [512,512]
    const float* b_proj = (const float*)d_in[3];   // [512]
    float* out = (float*)d_out;                    // [8,4096,512] f32

    char* ws = (char*)d_ws;
    size_t off = 0;
    auto carve = [&](size_t bytes) {
        char* p = ws + off;
        off += (bytes + 255) & ~(size_t)255;
        return p;
    };
    bf16* x_bf    = (bf16*)carve((size_t)32768 * 512 * 2);
    bf16* wqkv_bf = (bf16*)carve((size_t)1536 * 512 * 2);
    bf16* wproj_bf= (bf16*)carve((size_t)512 * 512 * 2);
    bf16* q_bf    = (bf16*)carve((size_t)64 * 4096 * 64 * 2);
    bf16* k_bf    = (bf16*)carve((size_t)64 * 4096 * 64 * 2);
    bf16* v_bf    = (bf16*)carve((size_t)64 * 4096 * 64 * 2);
    bf16* kp_bf   = (bf16*)carve((size_t)64 * 288 * 64 * 2);
    bf16* vp_bf   = (bf16*)carve((size_t)64 * 288 * 64 * 2);
    bf16* ao_bf   = x_bf;   // alias: x_bf fully consumed before attention

    // f32 -> bf16 conversions
    {
        int n = 32768 * 512;
        cvt_f32_bf16<<<n / 8 / 256, 256, 0, stream>>>(x, x_bf, n);
        n = 1536 * 512;
        cvt_f32_bf16<<<n / 8 / 256, 256, 0, stream>>>(w_qkv, wqkv_bf, n);
        n = 512 * 512;
        cvt_f32_bf16<<<n / 8 / 256, 256, 0, stream>>>(w_proj, wproj_bf, n);
    }

    // qkv projection  (M=32768, N=1536)
    gemm_qkv<<<dim3(32768 / 128, 1536 / 128), 256, GEMM_SMEM_BYTES, stream>>>(
        x_bf, wqkv_bf, q_bf, k_bf, v_bf);

    // PSP pooling of K and V
    psp_pool<<<64, 256, 0, stream>>>(k_bf, kp_bf);
    psp_pool<<<64, 256, 0, stream>>>(v_bf, vp_bf);

    // fused attention (64 bh x 64 query tiles of 64 rows)
    attn_fused<<<dim3(64, 4096 / 64), 128, ATTN_SMEM_BYTES, stream>>>(
        q_bf, kp_bf, vp_bf, ao_bf);

    // output projection (M=32768, N=512) + bias -> f32
    gemm_proj<<<dim3(32768 / 128, 512 / 128), 256, GEMM_SMEM_BYTES, stream>>>(
        ao_bf, wproj_bf, b_proj, out);
}
